// VisionEncoder_76209899700392
// MI455X (gfx1250) — matrix operand order repaired and checked by reading, not compiled, not run on hardware
//
#include <hip/hip_runtime.h>
#include <hip/hip_bf16.h>
#include <stdint.h>

// ---------------------------------------------------------------------------
// VQ nearest-codebook argmin on gfx1250 (MI455X).
//   dist = ||c||^2 - 2*x.c   (||x||^2 dropped: constant per row)
//   GEMM x.c^T via v_wmma_f32_16x16x32_bf16 with 3-term bf16 split
//   (hi*hi + lo*hi + hi*lo) for ~f32 accuracy.
//   Fast path: codebook pre-split once into ws as per-(chunk,kstep) 16KB LDS
//   images, streamed into LDS with global_load_async_to_lds_b128 (ASYNCcnt),
//   double-buffered under the WMMAs.
// ---------------------------------------------------------------------------

typedef __attribute__((ext_vector_type(16))) __bf16 v16bf;
typedef __attribute__((ext_vector_type(8)))  float  v8f;

#define NROWS   8192      // 32*256 feature rows
#define DDIM    384
#define KTOT    25000
#define MTILE   128       // rows per workgroup (8 waves x 16 rows)
#define NB      128       // codes per chunk (8 N-tiles of 16)
#define NCHUNK  196       // ceil(25000/128)
#define NCODE_PAD (NCHUNK * NB)   // 25088 (zero-padded codes)
#define KSTEPS  12        // 384 / 32
#define GRIDY   8         // K-dimension split across blocks
#define BLKBYTES 16384    // one (chunk,kstep) staged image: 8KB hi + 8KB lo

__device__ __forceinline__ v8f wmma_bf16(v16bf a, v16bf b, v8f c) {
  // (neg_a, A, neg_b, B, c_mod, C, reuse_a, reuse_b)
  return __builtin_amdgcn_wmma_f32_16x16x32_bf16(false, a, false, b,
                                                 (short)0, c, false, false);
}

// gfx1250 async global->LDS copy, 16B per lane, tracked by ASYNCcnt (GV mode).
__device__ __forceinline__ void async_ld_b128(uint32_t lds_off, const void* g) {
  asm volatile("global_load_async_to_lds_b128 %0, %1, off"
               :: "v"(lds_off), "v"(g) : "memory");
}
__device__ __forceinline__ void wait_async0() {
  asm volatile("s_wait_asynccnt 0x0" ::: "memory");
}

// ---- init running-min table -----------------------------------------------
__global__ void vq_init(unsigned long long* __restrict__ ws64) {
  int i = blockIdx.x * blockDim.x + threadIdx.x;
  if (i < NROWS) ws64[i] = 0xFFFFFFFFFFFFFFFFull;
}

// ---- codebook row norms ----------------------------------------------------
__global__ void vq_cnorm(const float* __restrict__ CB, float* __restrict__ cn) {
  const int lane = threadIdx.x & 31;
  const int wave = threadIdx.x >> 5;
  const int code = blockIdx.x * 8 + wave;
  if (code >= KTOT) return;
  const float* row = CB + (size_t)code * DDIM;
  float s = 0.0f;
#pragma unroll
  for (int j = 0; j < DDIM / 32; ++j) {
    float v = row[lane + j * 32];
    s = fmaf(v, v, s);
  }
#pragma unroll
  for (int m = 16; m >= 1; m >>= 1) s += __shfl_xor(s, m, 32);
  if (lane == 0) cn[code] = s;
}

// ---- one-time codebook split into per-(chunk,kstep) LDS images -------------
// Image layout (bf16): [0,4096)  hi  [khalf(2)][code(128)][16]
//                      [4096,8192) lo same indexing.
__global__ void vq_split(const float* __restrict__ CB,
                         unsigned short* __restrict__ SPu) {
  __bf16* SP = reinterpret_cast<__bf16*>(SPu);
  const int t = blockIdx.x * blockDim.x + threadIdx.x;   // (cg*12 + ks)*2 + kh
  if (t >= NCODE_PAD * KSTEPS * 2) return;
  const int kh    = t & 1;
  const int ks    = (t >> 1) % KSTEPS;
  const int cg    = t / (2 * KSTEPS);
  const int chunk = cg >> 7;
  const int cl    = cg & 127;
  float g[16];
  if (cg < KTOT) {
    const float* src = CB + (size_t)cg * DDIM + ks * 32 + kh * 16;
    const float4 p0 = *reinterpret_cast<const float4*>(src + 0);
    const float4 p1 = *reinterpret_cast<const float4*>(src + 4);
    const float4 p2 = *reinterpret_cast<const float4*>(src + 8);
    const float4 p3 = *reinterpret_cast<const float4*>(src + 12);
    g[0]=p0.x; g[1]=p0.y; g[2]=p0.z;  g[3]=p0.w;
    g[4]=p1.x; g[5]=p1.y; g[6]=p1.z;  g[7]=p1.w;
    g[8]=p2.x; g[9]=p2.y; g[10]=p2.z; g[11]=p2.w;
    g[12]=p3.x;g[13]=p3.y;g[14]=p3.z; g[15]=p3.w;
  } else {
#pragma unroll
    for (int e = 0; e < 16; ++e) g[e] = 0.0f;
  }
  __bf16* dh = SP + (size_t)(chunk * KSTEPS + ks) * 8192 + (kh * NB + cl) * 16;
  __bf16* dl = dh + 4096;
#pragma unroll
  for (int e = 0; e < 16; ++e) {
    __bf16 h = (__bf16)g[e];
    dh[e] = h;
    dl[e] = (__bf16)(g[e] - (float)h);
  }
}

// ---- shared helper: load & split this wave's A fragments -------------------
__device__ __forceinline__ void load_A(const float* __restrict__ X, int row,
                                       int kb, v16bf* Ah, v16bf* Al) {
  const float* xr = X + (size_t)row * DDIM;
#pragma unroll
  for (int ks = 0; ks < KSTEPS; ++ks) {
    float f[16];
    const float4 q0 = *reinterpret_cast<const float4*>(xr + ks * 32 + kb);
    const float4 q1 = *reinterpret_cast<const float4*>(xr + ks * 32 + kb + 4);
    const float4 q2 = *reinterpret_cast<const float4*>(xr + ks * 32 + kb + 8);
    const float4 q3 = *reinterpret_cast<const float4*>(xr + ks * 32 + kb + 12);
    f[0]=q0.x; f[1]=q0.y; f[2]=q0.z;  f[3]=q0.w;
    f[4]=q1.x; f[5]=q1.y; f[6]=q1.z;  f[7]=q1.w;
    f[8]=q2.x; f[9]=q2.y; f[10]=q2.z; f[11]=q2.w;
    f[12]=q3.x;f[13]=q3.y;f[14]=q3.z; f[15]=q3.w;
#pragma unroll
    for (int e = 0; e < 16; ++e) {
      __bf16 h = (__bf16)f[e];
      Ah[ks][e] = h;
      Al[ks][e] = (__bf16)(f[e] - (float)h);
    }
  }
}

__device__ __forceinline__ void merge_out(float* bestD, int* bestI, int lane,
                                          int rowBase,
                                          unsigned long long* __restrict__ ws64) {
#pragma unroll
  for (int r = 0; r < 8; ++r) {
    float d = bestD[r];
    int   i = bestI[r];
#pragma unroll
    for (int m = 1; m <= 8; m <<= 1) {
      float od = __shfl_xor(d, m, 32);
      int   oi = __shfl_xor(i, m, 32);
      if (od < d || (od == d && oi < i)) { d = od; i = oi; }
    }
    if ((lane & 15) == 0) {
      unsigned fb  = __float_as_uint(d);
      unsigned key = (fb & 0x80000000u) ? ~fb : (fb | 0x80000000u);
      unsigned long long packed = ((unsigned long long)key << 32) | (unsigned)i;
      atomicMin(ws64 + rowBase + (lane >> 4) * 8 + r, packed);
    }
  }
}

// ---- fast main kernel: async double-buffered staging -----------------------
__global__ __launch_bounds__(256) void vq_main_async(
    const float* __restrict__ X, const unsigned short* __restrict__ SPu,
    const float* __restrict__ cnorm, unsigned long long* __restrict__ ws64) {
  __shared__ __align__(128) __bf16 sB[2][8192];   // 2 x 16KB images

  const __bf16* SP = reinterpret_cast<const __bf16*>(SPu);
  const int tid  = threadIdx.x;
  const int lane = tid & 31;
  const int wave = tid >> 5;
  const int rowBase = blockIdx.x * MTILE + wave * 16;
  const int kb   = (lane >> 4) * 16;

  v16bf Ah[KSTEPS], Al[KSTEPS];
  load_A(X, rowBase + (lane & 15), kb, Ah, Al);

  float bestD[8];
  int   bestI[8];
#pragma unroll
  for (int r = 0; r < 8; ++r) { bestD[r] = 3.4e38f; bestI[r] = 0; }

  const uint32_t myOff = (uint32_t)tid * 64;   // each thread copies 64B
  uint32_t bases[2] = { (uint32_t)(uintptr_t)(&sB[0][0]),
                        (uint32_t)(uintptr_t)(&sB[1][0]) };

  // preload first image into buffer 0
  {
    const char* src = (const char*)SP +
        (((size_t)blockIdx.y * KSTEPS) << 14) + myOff;
    uint32_t l = bases[0] + myOff;
    async_ld_b128(l +  0, src +  0);
    async_ld_b128(l + 16, src + 16);
    async_ld_b128(l + 32, src + 32);
    async_ld_b128(l + 48, src + 48);
  }

  int buf = 0;
  for (int chunk = blockIdx.y; chunk < NCHUNK; chunk += GRIDY) {
    const int cbase = chunk * NB;
    v8f acc[8];
#pragma unroll
    for (int t = 0; t < 8; ++t)
#pragma unroll
      for (int e = 0; e < 8; ++e) acc[t][e] = 0.0f;

    for (int ks = 0; ks < KSTEPS; ++ks) {
      wait_async0();        // current buffer's copies complete (own wave)
      __syncthreads();      // all waves' copies done + old-buffer reads done

      // issue next image into the other buffer (overlaps with WMMAs below)
      int nc = chunk, nk = ks + 1;
      if (nk == KSTEPS) { nc = chunk + GRIDY; nk = 0; }
      if (nc < NCHUNK) {
        const char* src = (const char*)SP +
            (((size_t)nc * KSTEPS + nk) << 14) + myOff;
        uint32_t l = bases[buf ^ 1] + myOff;
        async_ld_b128(l +  0, src +  0);
        async_ld_b128(l + 16, src + 16);
        async_ld_b128(l + 32, src + 32);
        async_ld_b128(l + 48, src + 48);
      }

      const v16bf ah = Ah[ks];
      const v16bf al = Al[ks];
      const __bf16* bph = &sB[buf][0] + ((lane >> 4) * NB) * 16;
      const __bf16* bpl = bph + 4096;
#pragma unroll
      for (int t = 0; t < 8; ++t) {
        const int ci = (t * 16 + (lane & 15)) * 16;
        v16bf bh = *reinterpret_cast<const v16bf*>(bph + ci);
        v16bf bl = *reinterpret_cast<const v16bf*>(bpl + ci);
        acc[t] = wmma_bf16(ah, bh, acc[t]);   // hi*hi
        acc[t] = wmma_bf16(al, bh, acc[t]);   // lo*hi
        acc[t] = wmma_bf16(ah, bl, acc[t]);   // hi*lo
      }
      buf ^= 1;
    }

#pragma unroll
    for (int t = 0; t < 8; ++t) {
      const int code = cbase + t * 16 + (lane & 15);
      const float cn = (code < KTOT) ? cnorm[code] : 3.4e38f;
#pragma unroll
      for (int r = 0; r < 8; ++r) {
        const float dd = __builtin_fmaf(-2.0f, acc[t][r], cn);
        if (dd < bestD[r]) { bestD[r] = dd; bestI[r] = code; }
      }
    }
  }
  merge_out(bestD, bestI, lane, rowBase, ws64);
}

// ---- fallback main kernel: inline f32->bf16 staging (small ws) -------------
__global__ __launch_bounds__(256) void vq_main(
    const float* __restrict__ X, const float* __restrict__ CB,
    const float* __restrict__ cnorm, unsigned long long* __restrict__ ws64) {
  __shared__ __align__(32) __bf16 sBh[2 * NB * 16];
  __shared__ __align__(32) __bf16 sBl[2 * NB * 16];

  const int tid  = threadIdx.x;
  const int lane = tid & 31;
  const int wave = tid >> 5;
  const int rowBase = blockIdx.x * MTILE + wave * 16;
  const int kb   = (lane >> 4) * 16;

  v16bf Ah[KSTEPS], Al[KSTEPS];
  load_A(X, rowBase + (lane & 15), kb, Ah, Al);

  float bestD[8];
  int   bestI[8];
#pragma unroll
  for (int r = 0; r < 8; ++r) { bestD[r] = 3.4e38f; bestI[r] = 0; }

  const int stageCode = tid >> 1;
  const int stageHalf = tid & 1;
  __bf16* dstH = sBh + (stageHalf * NB + stageCode) * 16;
  __bf16* dstL = sBl + (stageHalf * NB + stageCode) * 16;

  for (int chunk = blockIdx.y; chunk < NCHUNK; chunk += GRIDY) {
    const int cbase = chunk * NB;
    v8f acc[8];
#pragma unroll
    for (int t = 0; t < 8; ++t)
#pragma unroll
      for (int e = 0; e < 8; ++e) acc[t][e] = 0.0f;

    const int   scode = cbase + stageCode;
    const bool  sval  = (scode < KTOT);
    const float* srow = CB + (size_t)scode * DDIM + stageHalf * 16;

    for (int ks = 0; ks < KSTEPS; ++ks) {
      __syncthreads();
      float g[16];
      if (sval) {
        const float4 p0 = *reinterpret_cast<const float4*>(srow + ks * 32 + 0);
        const float4 p1 = *reinterpret_cast<const float4*>(srow + ks * 32 + 4);
        const float4 p2 = *reinterpret_cast<const float4*>(srow + ks * 32 + 8);
        const float4 p3 = *reinterpret_cast<const float4*>(srow + ks * 32 + 12);
        g[0]=p0.x; g[1]=p0.y; g[2]=p0.z;  g[3]=p0.w;
        g[4]=p1.x; g[5]=p1.y; g[6]=p1.z;  g[7]=p1.w;
        g[8]=p2.x; g[9]=p2.y; g[10]=p2.z; g[11]=p2.w;
        g[12]=p3.x;g[13]=p3.y;g[14]=p3.z; g[15]=p3.w;
        if (ks + 1 < KSTEPS) __builtin_prefetch(srow + (ks + 1) * 32, 0, 0);
      } else {
#pragma unroll
        for (int e = 0; e < 16; ++e) g[e] = 0.0f;
      }
#pragma unroll
      for (int e = 0; e < 16; ++e) {
        __bf16 h = (__bf16)g[e];
        dstH[e] = h;
        dstL[e] = (__bf16)(g[e] - (float)h);
      }
      __syncthreads();

      const v16bf ah = Ah[ks];
      const v16bf al = Al[ks];
      const __bf16* bph = sBh + ((lane >> 4) * NB) * 16;
      const __bf16* bpl = sBl + ((lane >> 4) * NB) * 16;
#pragma unroll
      for (int t = 0; t < 8; ++t) {
        const int ci = (t * 16 + (lane & 15)) * 16;
        v16bf bh = *reinterpret_cast<const v16bf*>(bph + ci);
        v16bf bl = *reinterpret_cast<const v16bf*>(bpl + ci);
        acc[t] = wmma_bf16(ah, bh, acc[t]);
        acc[t] = wmma_bf16(al, bh, acc[t]);
        acc[t] = wmma_bf16(ah, bl, acc[t]);
      }
    }

#pragma unroll
    for (int t = 0; t < 8; ++t) {
      const int code = cbase + t * 16 + (lane & 15);
      const float cn = (code < KTOT) ? cnorm[code] : 3.4e38f;
#pragma unroll
      for (int r = 0; r < 8; ++r) {
        const float dd = __builtin_fmaf(-2.0f, acc[t][r], cn);
        if (dd < bestD[r]) { bestD[r] = dd; bestI[r] = code; }
      }
    }
  }
  merge_out(bestD, bestI, lane, rowBase, ws64);
}

// ---- unpack winning index --------------------------------------------------
__global__ void vq_fin(const unsigned long long* __restrict__ ws64,
                       int* __restrict__ tok) {
  int i = blockIdx.x * blockDim.x + threadIdx.x;
  if (i < NROWS) tok[i] = (int)(unsigned)(ws64[i] & 0xFFFFFFFFull);
}

// ---------------------------------------------------------------------------
extern "C" void kernel_launch(void* const* d_in, const int* in_sizes, int n_in,
                              void* d_out, int out_size, void* d_ws, size_t ws_size,
                              hipStream_t stream) {
  const float* X  = (const float*)d_in[0];   // features [32,256,384] f32
  const float* CB = (const float*)d_in[1];   // codebook [25000,384] f32

  unsigned long long* ws64 = (unsigned long long*)d_ws;       // 8192 * 8B
  float* cn = (float*)((char*)d_ws + (size_t)NROWS * 8);      // 25000 * 4B

  int*   tok      = (int*)d_out;
  float* feat_out = (float*)d_out + NROWS;

  const size_t SPLIT_OFF   = 262144;
  const size_t SPLIT_BYTES = (size_t)NCHUNK * KSTEPS * BLKBYTES; // 38.5 MB

  vq_init <<<(NROWS + 255) / 256, 256, 0, stream>>>(ws64);
  vq_cnorm<<<(KTOT + 7) / 8,      256, 0, stream>>>(CB, cn);

  if (ws_size >= SPLIT_OFF + SPLIT_BYTES) {
    unsigned short* SP = (unsigned short*)((char*)d_ws + SPLIT_OFF);
    vq_split<<<(NCODE_PAD * KSTEPS * 2) / 256, 256, 0, stream>>>(CB, SP);
    vq_main_async<<<dim3(NROWS / MTILE, GRIDY), 256, 0, stream>>>(X, SP, cn, ws64);
  } else {
    vq_main<<<dim3(NROWS / MTILE, GRIDY), 256, 0, stream>>>(X, CB, cn, ws64);
  }

  vq_fin<<<(NROWS + 255) / 256, 256, 0, stream>>>(ws64, tok);

  hipMemcpyAsync(feat_out, X, (size_t)NROWS * DDIM * sizeof(float),
                 hipMemcpyDeviceToDevice, stream);
}
